// Encoder_84241488543927
// MI455X (gfx1250) — compile-verified, matrix operand order checked by reading, not run on hardware
//
#include <hip/hip_runtime.h>
#include <hip/hip_bf16.h>
#include <stdint.h>

#define NU 4000
#define NI 4000
#define RR 5
#define MM 256
#define OUTD 75
#define SIDE 64
#define FDIM 128
#define KSPLIT 2016   // 63*32 ; second half 1984 = 62*32

typedef __attribute__((ext_vector_type(16))) __bf16 v16bf;
typedef __attribute__((ext_vector_type(8)))  float  v8f;

// ---------- helpers ----------
__device__ __forceinline__ unsigned short f32_to_bf16_bits(float f) {
    union { float f; unsigned int u; } c; c.f = f;
    unsigned int lsb = (c.u >> 16) & 1u;
    return (unsigned short)((c.u + 0x7FFFu + lsb) >> 16);   // RNE
}

union FragA { unsigned short s[16]; v16bf v; };
union FragB { int4 q[2];            v16bf v; };

// ---------- K1a: per-row counts -> ru = rsqrt(Nu) (0 if empty row) ----------
__global__ __launch_bounds__(256) void row_count_kernel(const int* __restrict__ adj,
                                                        float* __restrict__ ru) {
    __shared__ int red[256];
    const int u = blockIdx.x;
    int cnt = 0;
    for (int i = threadIdx.x; i < NI; i += 256) {
        int v = adj[u * NI + i];
        cnt += (v >= 1 && v <= RR) ? 1 : 0;
    }
    red[threadIdx.x] = cnt; __syncthreads();
    for (int s = 128; s > 0; s >>= 1) {
        if (threadIdx.x < s) red[threadIdx.x] += red[threadIdx.x + s];
        __syncthreads();
    }
    if (threadIdx.x == 0) {
        int n = red[0];
        ru[u] = (n > 0) ? rsqrtf((float)n) : 0.0f;
    }
}

// ---------- K1b: per-column counts -> rv = rsqrt(Nv) ----------
__global__ __launch_bounds__(256) void col_count_kernel(const int* __restrict__ adj,
                                                        float* __restrict__ rv) {
    const int i = blockIdx.x * 256 + threadIdx.x;
    if (i >= NI) return;
    int cnt = 0;
    for (int u = 0; u < NU; ++u) {
        int v = adj[u * NI + i];
        cnt += (v >= 1 && v <= RR) ? 1 : 0;
    }
    rv[i] = (cnt > 0) ? rsqrtf((float)cnt) : 0.0f;
}

// ---------- K3: pre-scale msg_W into bf16: Wus[r][m][u]=ru[u]*W, Wit[r][m][i]=rv[i]*W ----------
__global__ __launch_bounds__(256) void scale_w_kernel(const float* __restrict__ msgW,
                                                      const float* __restrict__ ru,
                                                      const float* __restrict__ rv,
                                                      unsigned short* __restrict__ wus,
                                                      unsigned short* __restrict__ wit) {
    const int idx = blockIdx.x * 256 + threadIdx.x;      // over RR*MM*(NU+NI) = 10,240,000
    const int x  = idx % (NU + NI);
    const int rm = idx / (NU + NI);
    const float w = msgW[idx];
    if (x < NU) wus[rm * NU + x]        = f32_to_bf16_bits(w * ru[x]);
    else        wit[rm * NI + (x - NU)] = f32_to_bf16_bits(w * rv[x - NU]);
}

// ---------- K4: Hu partials. wave = 4 tiles (16 rows x 64 cols); grid=(250, 2 K-halves).
// Hp[khalf][u][m] = sum_r sum_{i in Khalf} ind(adj[u,i]==r+1) * Wit[r][m][i]
__global__ __launch_bounds__(128) void hu_wmma_kernel(const int* __restrict__ adj,
                                                      const unsigned short* __restrict__ wit,
                                                      float* __restrict__ Hp) {
    const int lane  = threadIdx.x & 31;
    const int wave  = threadIdx.x >> 5;      // 0..3 = m-group (64 cols)
    const int u0    = blockIdx.x * 16;
    const int m0    = wave * 64;
    const int lrow  = lane & 15;
    const int khalf = lane >> 4;
    const int kbeg  = (blockIdx.y == 0) ? 0 : KSPLIT;
    const int kend  = (blockIdx.y == 0) ? KSPLIT : NI;

    const int* arow = adj + (size_t)(u0 + lrow) * NI;
    const unsigned short* brow = wit + (size_t)(m0 + lrow) * NI;

    v8f acc0 = {}, acc1 = {}, acc2 = {}, acc3 = {};
    for (int k0 = kbeg; k0 < kend; k0 += 32) {
        const int base = k0 + khalf * 8;
        union { int4 q[4]; int a[16]; } A;
        A.q[0] = *(const int4*)(arow + base);
        A.q[1] = *(const int4*)(arow + base + 4);
        A.q[2] = *(const int4*)(arow + base + 16);
        A.q[3] = *(const int4*)(arow + base + 20);
#pragma unroll
        for (int r = 0; r < RR; ++r) {
            FragA Af;
#pragma unroll
            for (int j = 0; j < 16; ++j)
                Af.s[j] = (A.a[j] == r + 1) ? (unsigned short)0x3F80 : (unsigned short)0;
            const unsigned short* bp = brow + (size_t)r * (MM * NI) + base;
            FragB B0, B1, B2, B3;
            B0.q[0] = *(const int4*)(bp);              B0.q[1] = *(const int4*)(bp + 16);
            B1.q[0] = *(const int4*)(bp + 16 * NI);    B1.q[1] = *(const int4*)(bp + 16 * NI + 16);
            B2.q[0] = *(const int4*)(bp + 32 * NI);    B2.q[1] = *(const int4*)(bp + 32 * NI + 16);
            B3.q[0] = *(const int4*)(bp + 48 * NI);    B3.q[1] = *(const int4*)(bp + 48 * NI + 16);
            acc0 = __builtin_amdgcn_wmma_f32_16x16x32_bf16(false, Af.v, false, B0.v, (short)0, acc0, false, false);
            acc1 = __builtin_amdgcn_wmma_f32_16x16x32_bf16(false, Af.v, false, B1.v, (short)0, acc1, false, false);
            acc2 = __builtin_amdgcn_wmma_f32_16x16x32_bf16(false, Af.v, false, B2.v, (short)0, acc2, false, false);
            acc3 = __builtin_amdgcn_wmma_f32_16x16x32_bf16(false, Af.v, false, B3.v, (short)0, acc3, false, false);
        }
    }
    float* out = Hp + (size_t)blockIdx.y * NU * MM;
#pragma unroll
    for (int j = 0; j < 8; ++j) {
        const int uo = u0 + 8 * khalf + j;
        float* o = out + (size_t)uo * MM + m0 + (lane & 15);
        o[0]  = acc0[j];
        o[16] = acc1[j];
        o[32] = acc2[j];
        o[48] = acc3[j];
    }
}

// ---------- K5: Hv partials; adj transposed via LDS tile (32u x 16i, padded 17). ----------
__global__ __launch_bounds__(128) void hv_wmma_kernel(const int* __restrict__ adj,
                                                      const unsigned short* __restrict__ wus,
                                                      float* __restrict__ Hp) {
    __shared__ int tileA[32][17];

    const int lane  = threadIdx.x & 31;
    const int wave  = threadIdx.x >> 5;      // 0..3 = m-group
    const int i0    = blockIdx.x * 16;
    const int m0    = wave * 64;
    const int lrow  = lane & 15;
    const int khalf = lane >> 4;
    const int kbeg  = (blockIdx.y == 0) ? 0 : KSPLIT;
    const int kend  = (blockIdx.y == 0) ? KSPLIT : NU;

    const unsigned short* brow = wus + (size_t)(m0 + lrow) * NU;

    const int t = threadIdx.x;               // 128 threads load 4 of 512 ints each
    v8f acc0 = {}, acc1 = {}, acc2 = {}, acc3 = {};
    for (int k0 = kbeg; k0 < kend; k0 += 32) {
#pragma unroll
        for (int c = 0; c < 4; ++c) {
            const int e  = t + c * 128;
            const int lk = e >> 4, li = e & 15;
            tileA[lk][li] = adj[(size_t)(k0 + lk) * NI + i0 + li];
        }
        __syncthreads();

        int a[16];
#pragma unroll
        for (int j = 0; j < 8; ++j) {
            a[j]     = tileA[khalf * 8 + j][lrow];
            a[8 + j] = tileA[16 + khalf * 8 + j][lrow];
        }
        const int base = k0 + khalf * 8;
#pragma unroll
        for (int r = 0; r < RR; ++r) {
            FragA Af;
#pragma unroll
            for (int j = 0; j < 16; ++j)
                Af.s[j] = (a[j] == r + 1) ? (unsigned short)0x3F80 : (unsigned short)0;
            const unsigned short* bp = brow + (size_t)r * (MM * NU) + base;
            FragB B0, B1, B2, B3;
            B0.q[0] = *(const int4*)(bp);              B0.q[1] = *(const int4*)(bp + 16);
            B1.q[0] = *(const int4*)(bp + 16 * NU);    B1.q[1] = *(const int4*)(bp + 16 * NU + 16);
            B2.q[0] = *(const int4*)(bp + 32 * NU);    B2.q[1] = *(const int4*)(bp + 32 * NU + 16);
            B3.q[0] = *(const int4*)(bp + 48 * NU);    B3.q[1] = *(const int4*)(bp + 48 * NU + 16);
            acc0 = __builtin_amdgcn_wmma_f32_16x16x32_bf16(false, Af.v, false, B0.v, (short)0, acc0, false, false);
            acc1 = __builtin_amdgcn_wmma_f32_16x16x32_bf16(false, Af.v, false, B1.v, (short)0, acc1, false, false);
            acc2 = __builtin_amdgcn_wmma_f32_16x16x32_bf16(false, Af.v, false, B2.v, (short)0, acc2, false, false);
            acc3 = __builtin_amdgcn_wmma_f32_16x16x32_bf16(false, Af.v, false, B3.v, (short)0, acc3, false, false);
        }
        __syncthreads();
    }
    float* out = Hp + (size_t)blockIdx.y * NI * MM;
#pragma unroll
    for (int j = 0; j < 8; ++j) {
        const int io = i0 + 8 * khalf + j;
        float* o = out + (size_t)io * MM + m0 + (lane & 15);
        o[0]  = acc0[j];
        o[16] = acc1[j];
        o[32] = acc2[j];
        o[48] = acc3[j];
    }
}

// ---------- K5b: combine K-half partials, scale by r[], relu ----------
__global__ __launch_bounds__(256) void h_finish_kernel(const float* __restrict__ Hp,
                                                       const float* __restrict__ rscale,
                                                       float* __restrict__ H) {
    const int idx = blockIdx.x * 256 + threadIdx.x;    // over 4000*256
    const int u = idx >> 8;
    const float s = rscale[u];
    H[idx] = fmaxf((Hp[idx] + Hp[(size_t)NU * MM + idx]) * s, 0.0f);
}

// ---------- K6: F = relu(feat @ W1^T + b1) ----------
__global__ __launch_bounds__(256) void side_mlp_kernel(const float* __restrict__ feat,
                                                       const float* __restrict__ W1,
                                                       const float* __restrict__ b1,
                                                       float* __restrict__ F, int n) {
    const int idx = blockIdx.x * 256 + threadIdx.x;
    if (idx >= n * SIDE) return;
    const int s = idx & (SIDE - 1);
    const int u = idx >> 6;
    const float* f = feat + (size_t)u * FDIM;
    const float* w = W1 + (size_t)s * FDIM;
    float acc = b1[s];
#pragma unroll 8
    for (int k = 0; k < FDIM; ++k) acc = fmaf(f[k], w[k], acc);
    F[idx] = fmaxf(acc, 0.0f);
}

// ---------- K7: out[u][o] = relu( H[u]·denseW[o] + F[u]·W2[o] ) ----------
__global__ __launch_bounds__(256) void out_proj_kernel(const float* __restrict__ H,
                                                       const float* __restrict__ F,
                                                       const float* __restrict__ denseW,
                                                       const float* __restrict__ W2,
                                                       float* __restrict__ out, int n) {
    const int idx = blockIdx.x * 256 + threadIdx.x;
    if (idx >= n * OUTD) return;
    const int o = idx % OUTD;
    const int u = idx / OUTD;
    const float* h  = H + (size_t)u * MM;
    const float* dw = denseW + (size_t)o * MM;
    float acc = 0.0f;
#pragma unroll 8
    for (int m = 0; m < MM; ++m) acc = fmaf(h[m], dw[m], acc);
    const float* f  = F + (size_t)u * SIDE;
    const float* w2 = W2 + (size_t)o * SIDE;
#pragma unroll 8
    for (int s = 0; s < SIDE; ++s) acc = fmaf(f[s], w2[s], acc);
    out[idx] = fmaxf(acc, 0.0f);
}

// ---------- launch ----------
extern "C" void kernel_launch(void* const* d_in, const int* in_sizes, int n_in,
                              void* d_out, int out_size, void* d_ws, size_t ws_size,
                              hipStream_t stream) {
    const int*   adj     = (const int*)  d_in[0];
    const float* u_side  = (const float*)d_in[1];
    const float* v_side  = (const float*)d_in[2];
    const float* msgW    = (const float*)d_in[3];
    const float* denseW  = (const float*)d_in[4];
    const float* u_W1    = (const float*)d_in[5];
    const float* u_b1    = (const float*)d_in[6];
    const float* u_W2    = (const float*)d_in[7];
    const float* v_W1    = (const float*)d_in[8];
    const float* v_b1    = (const float*)d_in[9];
    const float* v_W2    = (const float*)d_in[10];
    // d_in[11] (isNegative) unused by the reference.

    char* ws = (char*)d_ws;
    float*          ru  = (float*)         (ws + 0);
    float*          rv  = (float*)         (ws + 16384);
    unsigned short* wus = (unsigned short*)(ws + 32768);                 // 10,240,000 B
    unsigned short* wit = (unsigned short*)(ws + 32768 + 10240000);      // 10,240,000 B
    float*          Hu  = (float*)         (ws + 32768 + 20480000);      //  4,096,000 B
    float*          Hv  = (float*)         (ws + 32768 + 24576000);      //  4,096,000 B
    float*          Fu  = (float*)         (ws + 32768 + 28672000);      //  1,024,000 B
    float*          Fv  = (float*)         (ws + 32768 + 29696000);      //  1,024,000 B
    float*          Hup = (float*)         (ws + 32768 + 30720000);      //  8,192,000 B (2 K-halves)
    float*          Hvp = (float*)         (ws + 32768 + 38912000);      //  8,192,000 B

    float* U = (float*)d_out;
    float* V = U + (size_t)NU * OUTD;

    row_count_kernel<<<NU, 256, 0, stream>>>(adj, ru);
    col_count_kernel<<<(NI + 255) / 256, 256, 0, stream>>>(adj, rv);
    scale_w_kernel<<<(RR * MM * (NU + NI)) / 256, 256, 0, stream>>>(msgW, ru, rv, wus, wit);

    hu_wmma_kernel<<<dim3(NU / 16, 2), 128, 0, stream>>>(adj, wit, Hup);
    hv_wmma_kernel<<<dim3(NI / 16, 2), 128, 0, stream>>>(adj, wus, Hvp);
    h_finish_kernel<<<(NU * MM) / 256, 256, 0, stream>>>(Hup, ru, Hu);
    h_finish_kernel<<<(NI * MM) / 256, 256, 0, stream>>>(Hvp, rv, Hv);

    side_mlp_kernel<<<(NU * SIDE + 255) / 256, 256, 0, stream>>>(u_side, u_W1, u_b1, Fu, NU);
    side_mlp_kernel<<<(NI * SIDE + 255) / 256, 256, 0, stream>>>(v_side, v_W1, v_b1, Fv, NI);

    out_proj_kernel<<<(NU * OUTD + 255) / 256, 256, 0, stream>>>(Hu, Fu, denseW, u_W2, U, NU);
    out_proj_kernel<<<(NI * OUTD + 255) / 256, 256, 0, stream>>>(Hv, Fv, denseW, v_W2, V, NI);
}
